// TreatmentEffectEncoder_75325136437881
// MI455X (gfx1250) — matrix operand order repaired and checked by reading, not compiled, not run on hardware
//
#include <hip/hip_runtime.h>
#include <hip/hip_bf16.h>

// TreatmentEffectEncoder for MI455X (gfx1250, wave32, WMMA).
// One workgroup (8 waves) per batch row. f16 WMMA GEMMs (f32 accumulate),
// f32 LayerNorm/GELU/pooling. Weights + embedding table staged in LDS.

typedef __attribute__((ext_vector_type(16))) _Float16 v16h;
typedef __attribute__((ext_vector_type(8)))  _Float16 v8h;
typedef __attribute__((ext_vector_type(8)))  float    v8f;

#define EMBD   128
#define TDIM   32
#define KDIM   160           // EMBD + TDIM
#define LSEQ   200
#define NTYPES 50
#define MT     16            // tokens per chunk (one WMMA M-tile)
#define NCHUNK 13            // ceil(200/16)

// LDS strides (elements), padded for bank-conflict-free rows & 16B alignment
#define COMB_S 168           // 16x160 f16 A1 tile
#define W1T_S  168           // 128x160 f16 W1^T (row = out feature n)
#define W2T_S  136           // 128x128 f16 W2^T
#define A2_S   136           // 16x128 f16 A2 tile
#define H_S    132           // 16x128 f32 intermediate / scratch

#define N_F16  (NTYPES*EMBD + EMBD*W1T_S + EMBD*W2T_S + MT*COMB_S + MT*A2_S)
#define N_F32  (MT*H_S + 6*EMBD + 4*TDIM)
#define SMEM_BYTES (size_t)(N_F16*2 + N_F32*4 + MT*4)

__device__ __forceinline__ float gelu_exact(float y) {
    return 0.5f * y * (1.0f + erff(y * 0.70710678118654752f));
}

// A fragment, 16-bit 16x32: lane m holds row M=m(&15);
// lanes 0-15: K = kb*32 + 0..15 contiguous; lanes 16-31: K = kb*32 + 16..31.
__device__ __forceinline__ v16h load_frag_a(const _Float16* row, int kbase) {
    v8h lo = *(const v8h*)(row + kbase);
    v8h hi = *(const v8h*)(row + kbase + 8);
    return __builtin_shufflevector(lo, hi, 0,1,2,3,4,5,6,7,8,9,10,11,12,13,14,15);
}

// B fragment (contiguous k order, same as load_frag_a):
// lane n holds column N=n(&15); lanes 0-15: K = kb*32 + 0..15 contiguous,
// lanes 16-31: K = kb*32 + 16..31.  Stored n-major (W^T) so runs are contiguous.
__device__ __forceinline__ v16h load_frag_b(const _Float16* row, int kbase) {
    v8h lo = *(const v8h*)(row + kbase);
    v8h hi = *(const v8h*)(row + kbase + 8);
    return __builtin_shufflevector(lo, hi, 0,1,2,3,4,5,6,7,8,9,10,11,12,13,14,15);
}

__global__ __launch_bounds__(256) void tee_encoder_kernel(
    const int*   __restrict__ ids,   const float* __restrict__ dates,
    const int*   __restrict__ lengths, const float* __restrict__ emb,
    const float* __restrict__ tp_w,  const float* __restrict__ tp_b,
    const float* __restrict__ tp_lnw,const float* __restrict__ tp_lnb,
    const float* __restrict__ w1,    const float* __restrict__ b1,
    const float* __restrict__ ln1w,  const float* __restrict__ ln1b,
    const float* __restrict__ w2,    const float* __restrict__ b2,
    const float* __restrict__ ln2w,  const float* __restrict__ ln2b,
    float* __restrict__ out)
{
    extern __shared__ char smem[];
    _Float16* s_emb  = (_Float16*)smem;                 // [50][128]
    _Float16* s_w1t  = s_emb  + NTYPES*EMBD;            // [128][W1T_S]
    _Float16* s_w2t  = s_w1t  + EMBD*W1T_S;             // [128][W2T_S]
    _Float16* s_comb = s_w2t  + EMBD*W2T_S;             // [16][COMB_S]
    _Float16* s_a2   = s_comb + MT*COMB_S;              // [16][A2_S]
    float*    s_h    = (float*)(s_a2 + MT*A2_S);        // [16][H_S]
    float*    s_b1v  = s_h   + MT*H_S;
    float*    s_l1w  = s_b1v + EMBD;
    float*    s_l1b  = s_l1w + EMBD;
    float*    s_b2v  = s_l1b + EMBD;
    float*    s_l2w  = s_b2v + EMBD;
    float*    s_l2b  = s_l2w + EMBD;
    float*    s_tpw  = s_l2b + EMBD;
    float*    s_tpb  = s_tpw + TDIM;
    float*    s_tlw  = s_tpb + TDIM;
    float*    s_tlb  = s_tlw + TDIM;
    int*      s_idv  = (int*)(s_tlb + TDIM);

    const int tid = threadIdx.x;
    const int b   = blockIdx.x;
    int len = lengths[b];
    len = (len < 0) ? 0 : (len > LSEQ ? LSEQ : len);

    // ---- stage embedding table, weights (transposed, f16), params ----
    for (int i = tid; i < NTYPES*EMBD; i += 256) s_emb[i] = (_Float16)emb[i];
    for (int i = tid; i < KDIM*EMBD; i += 256) {
        int k = i >> 7, n = i & 127;                    // w1[k][n] -> w1t[n][k]
        s_w1t[n*W1T_S + k] = (_Float16)w1[i];
    }
    for (int i = tid; i < EMBD*EMBD; i += 256) {
        int k = i >> 7, n = i & 127;
        s_w2t[n*W2T_S + k] = (_Float16)w2[i];
    }
    if (tid < EMBD) {
        s_b1v[tid] = b1[tid]; s_l1w[tid] = ln1w[tid]; s_l1b[tid] = ln1b[tid];
        s_b2v[tid] = b2[tid]; s_l2w[tid] = ln2w[tid]; s_l2b[tid] = ln2b[tid];
    }
    if (tid < TDIM) {
        s_tpw[tid] = tp_w[tid]; s_tpb[tid] = tp_b[tid];
        s_tlw[tid] = tp_lnw[tid]; s_tlb[tid] = tp_lnb[tid];
    }

    const int lane = tid & 31;
    const int wv   = tid >> 5;           // 8 waves, each owns one 16-wide N-tile
    const int n0   = wv << 4;
    const int mrow = lane & 15;
    const int hi   = lane >> 4;
    const int tkn  = tid >> 4;           // token slot 0..15 (LN / pooling layout)
    const int fg   = tid & 15;           // feature group: feats fg*8 .. fg*8+7

    float pool[8];
    #pragma unroll
    for (int i = 0; i < 8; ++i) pool[i] = 0.0f;

    for (int c = 0; c < NCHUNK; ++c) {
        const int l0 = c * MT;
        __syncthreads();                                   // staging / prev chunk done

        // ---- A0: fetch token ids ----
        if (tid < MT) {
            int l = l0 + tid, id = -1;
            if (l < LSEQ) { id = ids[b*LSEQ + l]; if (id < 0 || id >= NTYPES) id = 0; }
            s_idv[tid] = id;
        }
        __syncthreads();

        // ---- A: build combined [16 x 160] f16 tile ----
        {   // embedding gather from LDS (cols 0..127), 8 f16 per thread
            int id = s_idv[tkn];
            v8h* dst = (v8h*)(s_comb + tkn*COMB_S + fg*8);
            if (id >= 0) *dst = *(const v8h*)(s_emb + id*EMBD + fg*8);
            else { v8h z = {0,0,0,0,0,0,0,0}; *dst = z; }
        }
        if (tid < MT) {   // temporal MLP: Linear(1->32) + LN + GELU (cols 128..159)
            int l = l0 + tid;
            _Float16* dst = s_comb + tid*COMB_S + EMBD;
            if (l < LSEQ) {
                float dn = dates[b*LSEQ + l] * (1.0f/1825.0f);
                float vb[TDIM]; float s = 0.f, q = 0.f;
                #pragma unroll
                for (int j = 0; j < TDIM; ++j) {
                    float x = dn*s_tpw[j] + s_tpb[j]; vb[j] = x; s += x; q += x*x;
                }
                float mean = s * (1.0f/TDIM);
                float var  = q * (1.0f/TDIM) - mean*mean;
                float rs   = rsqrtf(var + 1e-5f);
                #pragma unroll
                for (int j = 0; j < TDIM; ++j) {
                    float y = (vb[j]-mean)*rs*s_tlw[j] + s_tlb[j];
                    dst[j] = (_Float16)gelu_exact(y);
                }
            } else {
                #pragma unroll
                for (int j = 0; j < TDIM; ++j) dst[j] = (_Float16)0.0f;
            }
        }
        __syncthreads();

        // ---- GEMM1: [16x160] x [160x128], K = 5 WMMA steps ----
        {
            v8f acc = {};
            const _Float16* arow = s_comb + mrow*COMB_S + (hi ? 16 : 0);
            const _Float16* brow = s_w1t + (n0 + mrow)*W1T_S + (hi ? 16 : 0);
            #pragma unroll
            for (int kb = 0; kb < 5; ++kb) {
                v16h a  = load_frag_a(arow, kb*32);
                v16h bm = load_frag_b(brow, kb*32);
                acc = __builtin_amdgcn_wmma_f32_16x16x32_f16(
                        false, a, false, bm, (short)0, acc, false, false);
            }
            #pragma unroll
            for (int r = 0; r < 8; ++r) {                  // D layout: M=r+8*hi, N=lane&15
                int M = r + (hi ? 8 : 0);
                int N = n0 + mrow;
                s_h[M*H_S + N] = acc[r] + s_b1v[N];
            }
        }
        __syncthreads();

        // ---- LN1 + GELU -> f16 A2 tile ----
        {
            const float* hp = s_h + tkn*H_S + fg*8;
            float x[8]; float s = 0.f, q = 0.f;
            #pragma unroll
            for (int i = 0; i < 8; ++i) { x[i] = hp[i]; s += x[i]; q += x[i]*x[i]; }
            #pragma unroll
            for (int off = 8; off >= 1; off >>= 1) {       // reduce across 16-lane group
                s += __shfl_xor(s, off, 32); q += __shfl_xor(q, off, 32);
            }
            float mean = s * (1.0f/EMBD);
            float var  = q * (1.0f/EMBD) - mean*mean;
            float rs   = rsqrtf(var + 1e-5f);
            #pragma unroll
            for (int i = 0; i < 8; ++i) {
                int f = fg*8 + i;
                float y = (x[i]-mean)*rs*s_l1w[f] + s_l1b[f];
                s_a2[tkn*A2_S + f] = (_Float16)gelu_exact(y);
            }
        }
        __syncthreads();

        // ---- GEMM2: [16x128] x [128x128], K = 4 WMMA steps ----
        {
            v8f acc = {};
            const _Float16* arow = s_a2 + mrow*A2_S + (hi ? 16 : 0);
            const _Float16* brow = s_w2t + (n0 + mrow)*W2T_S + (hi ? 16 : 0);
            #pragma unroll
            for (int kb = 0; kb < 4; ++kb) {
                v16h a  = load_frag_a(arow, kb*32);
                v16h bm = load_frag_b(brow, kb*32);
                acc = __builtin_amdgcn_wmma_f32_16x16x32_f16(
                        false, a, false, bm, (short)0, acc, false, false);
            }
            #pragma unroll
            for (int r = 0; r < 8; ++r) {
                int M = r + (hi ? 8 : 0);
                int N = n0 + mrow;
                s_h[M*H_S + N] = acc[r] + s_b2v[N];
            }
        }
        __syncthreads();

        // ---- LN2 + masked mean-pool accumulate (registers) ----
        {
            const float* hp = s_h + tkn*H_S + fg*8;
            float x[8]; float s = 0.f, q = 0.f;
            #pragma unroll
            for (int i = 0; i < 8; ++i) { x[i] = hp[i]; s += x[i]; q += x[i]*x[i]; }
            #pragma unroll
            for (int off = 8; off >= 1; off >>= 1) {
                s += __shfl_xor(s, off, 32); q += __shfl_xor(q, off, 32);
            }
            float mean = s * (1.0f/EMBD);
            float var  = q * (1.0f/EMBD) - mean*mean;
            float rs   = rsqrtf(var + 1e-5f);
            bool valid = (l0 + tkn) < len;
            #pragma unroll
            for (int i = 0; i < 8; ++i) {
                int f = fg*8 + i;
                float y = (x[i]-mean)*rs*s_l2w[f] + s_l2b[f];
                if (valid) pool[i] += y;
            }
        }
    }

    // ---- final: reduce token-slot partials, divide by count, store ----
    __syncthreads();
    #pragma unroll
    for (int i = 0; i < 8; ++i) s_h[tkn*H_S + fg*8 + i] = pool[i];
    __syncthreads();
    if (tid < EMBD) {
        float s = 0.f;
        #pragma unroll
        for (int t = 0; t < MT; ++t) s += s_h[t*H_S + tid];
        out[b*EMBD + tid] = (len > 0) ? (s / (float)len) : 0.0f;
    }
}

extern "C" void kernel_launch(void* const* d_in, const int* in_sizes, int n_in,
                              void* d_out, int out_size, void* d_ws, size_t ws_size,
                              hipStream_t stream) {
    const int*   ids     = (const int*)  d_in[0];
    const float* dates   = (const float*)d_in[1];
    const int*   lengths = (const int*)  d_in[2];
    const float* emb     = (const float*)d_in[3];
    const float* tp_w    = (const float*)d_in[4];
    const float* tp_b    = (const float*)d_in[5];
    const float* tp_lnw  = (const float*)d_in[6];
    const float* tp_lnb  = (const float*)d_in[7];
    const float* w1      = (const float*)d_in[8];
    const float* b1      = (const float*)d_in[9];
    const float* ln1w    = (const float*)d_in[10];
    const float* ln1b    = (const float*)d_in[11];
    const float* w2      = (const float*)d_in[12];
    const float* b2      = (const float*)d_in[13];
    const float* ln2w    = (const float*)d_in[14];
    const float* ln2b    = (const float*)d_in[15];
    float* out = (float*)d_out;

    const int B = in_sizes[2];   // lengths has B elements (4096)
    tee_encoder_kernel<<<B, 256, SMEM_BYTES, stream>>>(
        ids, dates, lengths, emb, tp_w, tp_b, tp_lnw, tp_lnb,
        w1, b1, ln1w, ln1b, w2, b2, ln2w, ln2b, out);
}
